// MultiHeadAttention_38233798869101
// MI455X (gfx1250) — matrix operand order repaired and checked by reading, not compiled, not run on hardware
//
#include <hip/hip_runtime.h>
#include <hip/hip_bf16.h>

// ---- problem constants (match reference) ----
#define B_ 2
#define S_ 2048
#define D_ 1024
#define H_ 16
#define DK_ 64
#define MTOT (B_ * S_)          // 4096

typedef __attribute__((ext_vector_type(16))) __bf16 v16bf;
typedef __attribute__((ext_vector_type(8)))  __bf16 v8bf;
typedef __attribute__((ext_vector_type(8)))  float  v8f;

static __device__ __forceinline__ v16bf combine16(v8bf lo, v8bf hi) {
    v16bf r;
#pragma unroll
    for (int i = 0; i < 8; ++i) { r[i] = lo[i]; r[i + 8] = hi[i]; }
    return r;
}

// A-matrix fragment (16xK tile, row-major [M][K] in memory/LDS).
// ISA 16-bit A layout: lanes 0-15 -> row=lane, K {0..7,16..23}; lanes 16-31 -> row=lane-16, K {8..15,24..31}.
static __device__ __forceinline__ v16bf load_frag_a(const __bf16* base, int stride, int lane, int koff) {
    const int half = lane >> 4, r = lane & 15;
    const __bf16* p = base + (size_t)r * stride + koff + half * 8;
    return combine16(*(const v8bf*)p, *(const v8bf*)(p + 16));
}

// B-matrix fragment from B^T stored row-major [N][K].
// ISA 16-bit B layout: lanes 0-15 -> col=lane, K {0..7,16..23}; lanes 16-31 -> col=lane-16, K {8..15,24..31} (within 32-K chunk).
static __device__ __forceinline__ v16bf load_frag_b(const __bf16* base, int stride, int lane, int koff) {
    const int half = lane >> 4, n = lane & 15;
    const __bf16* p = base + (size_t)n * stride + koff + half * 8;
    return combine16(*(const v8bf*)p, *(const v8bf*)(p + 16));
}

static __device__ __forceinline__ v8f wmma_bf16(v16bf a, v16bf b, v8f c) {
    return __builtin_amdgcn_wmma_f32_16x16x32_bf16(false, a, false, b, (short)0, c, false, false);
}

static __device__ __forceinline__ float half_reduce_max(float v) {
    v = fmaxf(v, __shfl_xor(v, 1, 32));
    v = fmaxf(v, __shfl_xor(v, 2, 32));
    v = fmaxf(v, __shfl_xor(v, 4, 32));
    v = fmaxf(v, __shfl_xor(v, 8, 32));
    return v;
}
static __device__ __forceinline__ float half_reduce_sum(float v) {
    v += __shfl_xor(v, 1, 32);
    v += __shfl_xor(v, 2, 32);
    v += __shfl_xor(v, 4, 32);
    v += __shfl_xor(v, 8, 32);
    return v;
}

// fp32x8 -> bf16x8 convert + 16B LDS store
static __device__ __forceinline__ void cvt_store8(const float* __restrict__ g, __bf16* l) {
    float4 f0 = *(const float4*)g;
    float4 f1 = *(const float4*)(g + 4);
    v8bf v;
    v[0] = (__bf16)f0.x; v[1] = (__bf16)f0.y; v[2] = (__bf16)f0.z; v[3] = (__bf16)f0.w;
    v[4] = (__bf16)f1.x; v[5] = (__bf16)f1.y; v[6] = (__bf16)f1.z; v[7] = (__bf16)f1.w;
    *(v8bf*)l = v;
}

// async global(bf16)->LDS 16B copy, tracked by ASYNCcnt (CDNA5 async path)
static __device__ __forceinline__ void async_copy16(const __bf16* g, __bf16* l) {
    const unsigned lds = (unsigned)(uintptr_t)l;
    asm volatile("global_load_async_to_lds_b128 %0, %1, off"
                 :: "v"(lds), "v"(g) : "memory");
}
static __device__ __forceinline__ void wait_async0() {
    asm volatile("s_wait_asynccnt 0x0" ::: "memory");
}

// =====================================================================
// Kernel 1: fused QKV projection.  C[m][n] = x[m][:] . W[n][:]
// Block tile 128x128, K-step 32, double-buffered LDS, 8 waves,
// wave tile 32x64 (2 A-frags x 4 B-frags -> 8 WMMAs / K-step).
// Q,K stored [b][h][s][dk] bf16; V stored transposed [b][h][dk][s] bf16.
// =====================================================================
__global__ __launch_bounds__(256)
void mha_qkv_proj(const float* __restrict__ x,
                  const float* __restrict__ Wq,
                  const float* __restrict__ Wk,
                  const float* __restrict__ Wv,
                  __bf16* __restrict__ Qo,
                  __bf16* __restrict__ Ko,
                  __bf16* __restrict__ Vt) {
    __shared__ __align__(16) __bf16 sX[2][128 * 40];   // [128 m][32 k] pad->40
    __shared__ __align__(16) __bf16 sW[2][128 * 40];   // [128 n][32 k] pad->40

    const int tid  = threadIdx.x;
    const int lane = tid & 31;
    const int wave = tid >> 5;
    const int wm   = wave & 3;        // 4 waves along M (32 rows each)
    const int wn   = wave >> 2;       // 2 waves along N (64 cols each)
    const int m0   = blockIdx.x * 128;
    const int n0   = blockIdx.y * 128; // within 3*D
    const int proj = n0 >> 10;         // 0=Q,1=K,2=V (uniform per block)
    const int nW0  = n0 & 1023;
    const float* __restrict__ W = (proj == 0) ? Wq : (proj == 1) ? Wk : Wv;

    v8f acc[2][4];
#pragma unroll
    for (int i = 0; i < 2; ++i)
#pragma unroll
        for (int j = 0; j < 4; ++j)
#pragma unroll
            for (int r = 0; r < 8; ++r) acc[i][j][r] = 0.f;

    const int srow = tid >> 1;          // 0..127
    const int scol = (tid & 1) * 16;    // 0 / 16

    // stage K-tile kb into buffer buf
    auto stage = [&](int buf, int kb) {
        const float* px = x + (size_t)(m0 + srow) * D_ + kb + scol;
        cvt_store8(px,     &sX[buf][srow * 40 + scol]);
        cvt_store8(px + 8, &sX[buf][srow * 40 + scol + 8]);
        const float* pw = W + (size_t)(nW0 + srow) * D_ + kb + scol;
        cvt_store8(pw,     &sW[buf][srow * 40 + scol]);
        cvt_store8(pw + 8, &sW[buf][srow * 40 + scol + 8]);
    };

    stage(0, 0);
    for (int kb = 0; kb < D_; kb += 32) {
        __syncthreads();
        const int cur = (kb >> 5) & 1;
        if (kb + 32 < D_) stage(cur ^ 1, kb + 32);

        // preload all fragments into distinct registers so the 8 WMMAs
        // issue back-to-back behind a single partial dscnt wait
        v16bf a0 = load_frag_a(&sX[cur][(wm * 32 +  0) * 40], 40, lane, 0);
        v16bf a1 = load_frag_a(&sX[cur][(wm * 32 + 16) * 40], 40, lane, 0);
        v16bf b[4];
#pragma unroll
        for (int j = 0; j < 4; ++j)
            b[j] = load_frag_b(&sW[cur][(wn * 64 + j * 16) * 40], 40, lane, 0);
#pragma unroll
        for (int j = 0; j < 4; ++j) {
            acc[0][j] = wmma_bf16(a0, b[j], acc[0][j]);
            acc[1][j] = wmma_bf16(a1, b[j], acc[1][j]);
        }
    }

    // epilogue: scatter to Q / K / V^T in bf16
    const int half = lane >> 4;
#pragma unroll
    for (int i = 0; i < 2; ++i)
#pragma unroll
    for (int j = 0; j < 4; ++j) {
        const int ng   = n0 + wn * 64 + j * 16 + (lane & 15);
        const int dcol = ng & 1023;
        const int h    = dcol >> 6;
        const int dk   = dcol & 63;
#pragma unroll
        for (int r = 0; r < 8; ++r) {
            const int m  = m0 + wm * 32 + i * 16 + r + 8 * half;
            const int bb = m >> 11;          // /2048
            const int ss = m & 2047;
            const __bf16 v = (__bf16)acc[i][j][r];
            if (proj == 0)
                Qo[(((size_t)bb * H_ + h) * S_ + ss) * DK_ + dk] = v;
            else if (proj == 1)
                Ko[(((size_t)bb * H_ + h) * S_ + ss) * DK_ + dk] = v;
            else
                Vt[(((size_t)bb * H_ + h) * DK_ + dk) * S_ + ss] = v;
        }
    }
}

// =====================================================================
// Kernel 2: causal flash attention.
// Grid (S/64, B*H), block 128 (4 waves, 16 query rows each).
// K/V^T tiles staged with async global->LDS copies (ASYNCcnt),
// double-buffered so the DMA for tile j+1 overlaps compute on tile j.
// =====================================================================
__global__ __launch_bounds__(128)
void mha_flash_attn(const __bf16* __restrict__ Qg,
                    const __bf16* __restrict__ Kg,
                    const __bf16* __restrict__ Vtg,
                    __bf16* __restrict__ attn) {
    __shared__ __align__(16) __bf16 sK [2][64 * 72];   // [kv][dk] pad->72
    __shared__ __align__(16) __bf16 sVt[2][64 * 72];   // [dk][kv] pad->72
    __shared__ __align__(16) __bf16 sP [4 * 16 * 72];  // per-wave [q][kv]

    const int tid  = threadIdx.x;
    const int lane = tid & 31;
    const int wave = tid >> 5;
    const int half = lane >> 4;
    const int qb   = blockIdx.x;
    const int q0   = qb * 64;
    const int bh   = blockIdx.y;
    const int bIdx = bh >> 4;
    const int hIdx = bh & 15;

    const __bf16* __restrict__ Qb  = Qg  + (size_t)bh * S_ * DK_;
    const __bf16* __restrict__ Kb  = Kg  + (size_t)bh * S_ * DK_;
    const __bf16* __restrict__ Vtb = Vtg + (size_t)bh * DK_ * S_;

    // Q fragments for this wave's 16 rows (held in registers the whole kernel)
    const v16bf qa0 = load_frag_a(Qb + (size_t)(q0 + wave * 16) * DK_, DK_, lane, 0);
    const v16bf qa1 = load_frag_a(Qb + (size_t)(q0 + wave * 16) * DK_, DK_, lane, 32);

    v8f o[4];
    float mrow[8], lrow[8];
#pragma unroll
    for (int t = 0; t < 4; ++t)
#pragma unroll
        for (int r = 0; r < 8; ++r) o[t][r] = 0.f;
#pragma unroll
    for (int r = 0; r < 8; ++r) { mrow[r] = -1e30f; lrow[r] = 0.f; }

    const int srow = tid >> 1;            // 0..63
    const int sc0  = (tid & 1) * 32;      // 0 / 32

    // async-stage K tile [kv][dk] and V^T tile [dk][kv] for kv block j
    auto stage_async = [&](int buf, int j) {
        const int kv0 = j * 64;
#pragma unroll
        for (int cc = 0; cc < 4; ++cc) {
            async_copy16(Kb  + (size_t)(kv0 + srow) * DK_ + sc0 + cc * 8,
                         &sK[buf][srow * 72 + sc0 + cc * 8]);
            async_copy16(Vtb + (size_t)srow * S_ + kv0 + sc0 + cc * 8,
                         &sVt[buf][srow * 72 + sc0 + cc * 8]);
        }
    };

    stage_async(0, 0);
    for (int j = 0; j <= qb; ++j) {
        const int kv0 = j * 64;
        const int cur = j & 1;
        wait_async0();          // my async copies for tile j are done
        __syncthreads();        // everyone's are done
        if (j < qb) stage_async(cur ^ 1, j + 1);

        // ---- scores: 16 q rows x 64 kv (4 sub-tiles), B frags preloaded ----
        v8f sc[4];
        {
            v16bf kb[4];
#pragma unroll
            for (int t = 0; t < 4; ++t)
                kb[t] = load_frag_b(&sK[cur][t * 16 * 72], 72, lane, 0);
#pragma unroll
            for (int t = 0; t < 4; ++t) {
#pragma unroll
                for (int r = 0; r < 8; ++r) sc[t][r] = 0.f;
                sc[t] = wmma_bf16(qa0, kb[t], sc[t]);
            }
#pragma unroll
            for (int t = 0; t < 4; ++t)
                kb[t] = load_frag_b(&sK[cur][t * 16 * 72], 72, lane, 32);
#pragma unroll
            for (int t = 0; t < 4; ++t)
                sc[t] = wmma_bf16(qa1, kb[t], sc[t]);
        }

        // scale + causal mask (diagonal block only)
        const bool diag = (j == qb);
#pragma unroll
        for (int t = 0; t < 4; ++t)
#pragma unroll
            for (int r = 0; r < 8; ++r) {
                float v = sc[t][r] * 0.125f;  // 1/sqrt(64)
                if (diag) {
                    const int qi = q0 + wave * 16 + r + 8 * half;
                    const int ki = kv0 + t * 16 + (lane & 15);
                    if (ki > qi) v = -1e30f;
                }
                sc[t][r] = v;
            }

        // ---- online softmax over this kv tile ----
#pragma unroll
        for (int r = 0; r < 8; ++r) {
            float tm = sc[0][r];
            tm = fmaxf(tm, sc[1][r]); tm = fmaxf(tm, sc[2][r]); tm = fmaxf(tm, sc[3][r]);
            tm = half_reduce_max(tm);
            const float mnew  = fmaxf(mrow[r], tm);
            const float alpha = __expf(mrow[r] - mnew);
            mrow[r] = mnew;
            float s = 0.f;
#pragma unroll
            for (int t = 0; t < 4; ++t) {
                const float p = __expf(sc[t][r] - mnew);
                sc[t][r] = p;
                s += p;
            }
            s = half_reduce_sum(s);
            lrow[r] = lrow[r] * alpha + s;
#pragma unroll
            for (int t = 0; t < 4; ++t) o[t][r] *= alpha;
        }

        // ---- write P to LDS (accumulator layout -> row-major [q][kv]) ----
        __bf16* pw = &sP[wave * 16 * 72];
#pragma unroll
        for (int t = 0; t < 4; ++t)
#pragma unroll
            for (int r = 0; r < 8; ++r)
                pw[(r + 8 * half) * 72 + t * 16 + (lane & 15)] = (__bf16)sc[t][r];
        __syncthreads();

        // ---- O += P (16x64) @ V (64kv x 64dk), B^T = V^T tile in LDS ----
#pragma unroll
        for (int kc = 0; kc < 2; ++kc) {
            v16bf pa = load_frag_a(pw, 72, lane, kc * 32);
            v16bf vb[4];
#pragma unroll
            for (int t = 0; t < 4; ++t)
                vb[t] = load_frag_b(&sVt[cur][t * 16 * 72], 72, lane, kc * 32);
#pragma unroll
            for (int t = 0; t < 4; ++t)
                o[t] = wmma_bf16(pa, vb[t], o[t]);
        }
    }

    // ---- finalize: O /= l, write attn [b][s][h*64+dk] bf16 ----
#pragma unroll
    for (int r = 0; r < 8; ++r) {
        const float inv = 1.f / lrow[r];
        const int   ss  = q0 + wave * 16 + r + 8 * half;
#pragma unroll
        for (int t = 0; t < 4; ++t) {
            const int col = hIdx * DK_ + t * 16 + (lane & 15);
            attn[((size_t)bIdx * S_ + ss) * D_ + col] = (__bf16)(o[t][r] * inv);
        }
    }
}

// =====================================================================
// Kernel 3: output projection  out = attn @ Wp^T + bp   (fp32 out)
// Same 128x128 double-buffered tiling as kernel 1; A is already bf16.
// =====================================================================
__global__ __launch_bounds__(256)
void mha_out_proj(const __bf16* __restrict__ attn,
                  const float* __restrict__ Wp,
                  const float* __restrict__ bp,
                  float* __restrict__ out) {
    __shared__ __align__(16) __bf16 sX[2][128 * 40];
    __shared__ __align__(16) __bf16 sW[2][128 * 40];

    const int tid  = threadIdx.x;
    const int lane = tid & 31;
    const int wave = tid >> 5;
    const int wm   = wave & 3;
    const int wn   = wave >> 2;
    const int m0   = blockIdx.x * 128;
    const int n0   = blockIdx.y * 128;

    v8f acc[2][4];
#pragma unroll
    for (int i = 0; i < 2; ++i)
#pragma unroll
        for (int j = 0; j < 4; ++j)
#pragma unroll
            for (int r = 0; r < 8; ++r) acc[i][j][r] = 0.f;

    const int srow = tid >> 1;
    const int scol = (tid & 1) * 16;

    auto stage = [&](int buf, int kb) {
        // attn is bf16 already: straight 16B copies through VGPRs
        *(v8bf*)&sX[buf][srow * 40 + scol] =
            *(const v8bf*)(attn + (size_t)(m0 + srow) * D_ + kb + scol);
        *(v8bf*)&sX[buf][srow * 40 + scol + 8] =
            *(const v8bf*)(attn + (size_t)(m0 + srow) * D_ + kb + scol + 8);
        const float* pw = Wp + (size_t)(n0 + srow) * D_ + kb + scol;
        cvt_store8(pw,     &sW[buf][srow * 40 + scol]);
        cvt_store8(pw + 8, &sW[buf][srow * 40 + scol + 8]);
    };

    stage(0, 0);
    for (int kb = 0; kb < D_; kb += 32) {
        __syncthreads();
        const int cur = (kb >> 5) & 1;
        if (kb + 32 < D_) stage(cur ^ 1, kb + 32);

        v16bf a0 = load_frag_a(&sX[cur][(wm * 32 +  0) * 40], 40, lane, 0);
        v16bf a1 = load_frag_a(&sX[cur][(wm * 32 + 16) * 40], 40, lane, 0);
        v16bf b[4];
#pragma unroll
        for (int j = 0; j < 4; ++j)
            b[j] = load_frag_b(&sW[cur][(wn * 64 + j * 16) * 40], 40, lane, 0);
#pragma unroll
        for (int j = 0; j < 4; ++j) {
            acc[0][j] = wmma_bf16(a0, b[j], acc[0][j]);
            acc[1][j] = wmma_bf16(a1, b[j], acc[1][j]);
        }
    }

    const int half = lane >> 4;
#pragma unroll
    for (int i = 0; i < 2; ++i)
#pragma unroll
    for (int j = 0; j < 4; ++j) {
        const int ng   = n0 + wn * 64 + j * 16 + (lane & 15);
        const float bv = bp[ng];
#pragma unroll
        for (int r = 0; r < 8; ++r) {
            const int m = m0 + wm * 32 + i * 16 + r + 8 * half;
            out[(size_t)m * D_ + ng] = acc[i][j][r] + bv;
        }
    }
}

// =====================================================================
extern "C" void kernel_launch(void* const* d_in, const int* in_sizes, int n_in,
                              void* d_out, int out_size, void* d_ws, size_t ws_size,
                              hipStream_t stream) {
    const float* x  = (const float*)d_in[0];
    const float* Wq = (const float*)d_in[1];
    const float* Wk = (const float*)d_in[2];
    const float* Wv = (const float*)d_in[3];
    const float* Wp = (const float*)d_in[4];
    const float* bp = (const float*)d_in[5];
    float* out = (float*)d_out;

    const size_t headElems = (size_t)B_ * H_ * S_ * DK_;   // 4M elems
    char* ws = (char*)d_ws;
    __bf16* Q    = (__bf16*)(ws);
    __bf16* K    = (__bf16*)(ws + headElems * 2);
    __bf16* Vt   = (__bf16*)(ws + headElems * 4);
    __bf16* attn = (__bf16*)(ws + headElems * 6);

    dim3 g1(MTOT / 128, 3 * D_ / 128);
    mha_qkv_proj<<<g1, 256, 0, stream>>>(x, Wq, Wk, Wv, Q, K, Vt);

    dim3 g2(S_ / 64, B_ * H_);
    mha_flash_attn<<<g2, 128, 0, stream>>>(Q, K, Vt, attn);

    dim3 g3(MTOT / 128, D_ / 128);
    mha_out_proj<<<g3, 256, 0, stream>>>(attn, Wp, bp, out);
}
